// GNNStack_62758062129262
// MI455X (gfx1250) — matrix operand order repaired and checked by reading, not compiled, not run on hardware
//
#include <hip/hip_runtime.h>
#include <hip/hip_bf16.h>
#include <math.h>

typedef __attribute__((ext_vector_type(16))) _Float16 v16h;
typedef __attribute__((ext_vector_type(8)))  float    v8f;

#define HEADS 3
#define HID   64
#define FEAT  (HEADS * HID)   // 192

// ---------------- monotonic float<->uint key for atomic segment-max ----------
__device__ __forceinline__ unsigned fkey(float f) {
  unsigned u = __float_as_uint(f);
  return (u & 0x80000000u) ? ~u : (u | 0x80000000u);
}
__device__ __forceinline__ float fdecode(unsigned u) {
  unsigned b = (u & 0x80000000u) ? (u & 0x7FFFFFFFu) : ~u;
  return __uint_as_float(b);
}

// ---------------- generic zero fill ------------------------------------------
__global__ void fill0_kernel(float* __restrict__ p, int n) {
  int i = blockIdx.x * blockDim.x + threadIdx.x;
  if (i < n) p[i] = 0.0f;
}

// ---------------- WMMA GEMM: C[M,N] = A[M,K](f32) * B[K,N](f32) + bias -------
// K, N compile-time: all load offsets fold into 24-bit instruction immediates
// off a single base VGPR pair per stream; k-loop fully unrolls.
// One wave computes a 16(M) x 64(N) strip: 4 accumulators share the A fragment.
// REQUIRES K % 32 == 0. Edges handled by clamped loads + guarded stores.
template <int K, int N>
__global__ void wmma_gemm_bias(const float* __restrict__ A,
                               const float* __restrict__ B,
                               const float* __restrict__ bias,
                               float* __restrict__ C,
                               int M, int do_relu) {
  constexpr int tilesN = (N + 63) / 64;
  const int lane   = threadIdx.x & 31;
  const int wave   = (blockIdx.x * blockDim.x + threadIdx.x) >> 5;
  const int tilesM = (M + 15) >> 4;
  if (wave >= tilesM * tilesN) return;          // wave-uniform: EXEC stays full
  const int tm = (wave / tilesN) << 4;
  const int tn = (wave % tilesN) << 6;

  const int l  = lane & 15;
  const int hi = lane >> 4;                     // 0 or 1

  const int row = min(tm + l, M - 1);           // clamp: loads stay in-bounds
  const float* __restrict__ abase = A + (size_t)row * K + hi * 8;

  // one base pointer per output tile; all B offsets are compile-time
  const float* bbase[4];
#pragma unroll
  for (int t = 0; t < 4; ++t) {
    int cnt  = min(tn + t * 16 + l, N - 1);
    bbase[t] = B + (size_t)hi * 8 * N + cnt;
  }

  v8f c[4] = {{}, {}, {}, {}};

#pragma unroll
  for (int k0 = 0; k0 < K; k0 += 32) {
    // A fragment (16x32 f16 layout): 4 x b128 with immediate offsets
    float4 p0 = *(const float4*)(abase + k0);
    float4 p1 = *(const float4*)(abase + k0 + 4);
    float4 p2 = *(const float4*)(abase + k0 + 16);
    float4 p3 = *(const float4*)(abase + k0 + 20);
    v16h a;
    a[0]  = (_Float16)p0.x; a[1]  = (_Float16)p0.y;
    a[2]  = (_Float16)p0.z; a[3]  = (_Float16)p0.w;
    a[4]  = (_Float16)p1.x; a[5]  = (_Float16)p1.y;
    a[6]  = (_Float16)p1.z; a[7]  = (_Float16)p1.w;
    a[8]  = (_Float16)p2.x; a[9]  = (_Float16)p2.y;
    a[10] = (_Float16)p2.z; a[11] = (_Float16)p2.w;
    a[12] = (_Float16)p3.x; a[13] = (_Float16)p3.y;
    a[14] = (_Float16)p3.z; a[15] = (_Float16)p3.w;

#pragma unroll
    for (int t = 0; t < 4; ++t) {
      v16h b;
#pragma unroll
      for (int j = 0; j < 16; ++j)
        b[j] = (_Float16)bbase[t][(size_t)(k0 + j + (j & 8)) * N];   // immediate offset
      c[t] = __builtin_amdgcn_wmma_f32_16x16x32_f16(false, a, false, b,
                                                    (short)0, c[t], false, false);
    }
  }

  // C/D layout: VGPR r -> row tm + r + 8*hi, col tile_base + (lane&15)
  const bool fast = (tm + 16 <= M) && ((N & 63) == 0);   // second term constexpr
  float* __restrict__ crow = C + (size_t)(tm + hi * 8) * N;
  if (fast) {
#pragma unroll
    for (int t = 0; t < 4; ++t) {
      float bv = bias[tn + t * 16 + l];
#pragma unroll
      for (int r = 0; r < 8; ++r) {
        float v = c[t][r] + bv;
        if (do_relu) v = fmaxf(v, 0.0f);
        crow[(size_t)r * N + tn + t * 16 + l] = v;
      }
    }
  } else {
#pragma unroll
    for (int t = 0; t < 4; ++t) {
#pragma unroll
      for (int r = 0; r < 8; ++r) {
        int orow = tm + r + hi * 8;
        int ocol = tn + t * 16 + l;
        if (orow < M && ocol < N) {
          float v = c[t][r] + bias[ocol];
          if (do_relu) v = fmaxf(v, 0.0f);
          C[(size_t)orow * N + ocol] = v;
        }
      }
    }
  }
}

// ---------------- attention logits + leaky_relu + segment max ----------------
// alpha[e,h] = leaky_relu( H[dst]·att_i[h] + H[src]·att_j[h], 0.2 )
__global__ void edge_alpha_kernel(const float* __restrict__ H,
                                  const int* __restrict__ src,
                                  const int* __restrict__ dst,
                                  const float* __restrict__ att,  // (1,3,128)
                                  float* __restrict__ alpha,      // (E,3)
                                  unsigned* __restrict__ amaxu,   // (N,3) keys
                                  int E) {
  int e = blockIdx.x * blockDim.x + threadIdx.x;
  if (e >= E) return;
  int s = src[e], d = dst[e];
  const float4* Hs = (const float4*)(H + (size_t)s * FEAT);
  const float4* Hd = (const float4*)(H + (size_t)d * FEAT);
  const float4* at = (const float4*)att;
  __builtin_prefetch(Hs, 0, 3);   // global_prefetch_b8 path
  __builtin_prefetch(Hd, 0, 3);
#pragma unroll
  for (int h = 0; h < HEADS; ++h) {
    float acc = 0.0f;
#pragma unroll 4
    for (int q = 0; q < HID / 4; ++q) {
      float4 hd = Hd[h * 16 + q];
      float4 hs = Hs[h * 16 + q];
      float4 ai = at[h * 32 + q];
      float4 aj = at[h * 32 + 16 + q];
      acc += hd.x * ai.x + hd.y * ai.y + hd.z * ai.z + hd.w * ai.w;
      acc += hs.x * aj.x + hs.y * aj.y + hs.z * aj.z + hs.w * aj.w;
    }
    float a = (acc > 0.0f) ? acc : 0.2f * acc;
    alpha[e * HEADS + h] = a;
    atomicMax(amaxu + d * HEADS + h, fkey(a));
  }
}

// decode uint max-keys back to floats, in place
__global__ void amax_decode_kernel(unsigned* __restrict__ p, int n) {
  int i = blockIdx.x * blockDim.x + threadIdx.x;
  if (i < n) {
    unsigned u = p[i];
    ((float*)p)[i] = fdecode(u);
  }
}

// ex = exp(alpha - amax[dst]); alpha <- ex; denom[dst] += ex
__global__ void edge_exp_kernel(const float* __restrict__ amax,
                                const int* __restrict__ dst,
                                float* __restrict__ alpha,
                                float* __restrict__ denom,
                                int E) {
  int e = blockIdx.x * blockDim.x + threadIdx.x;
  if (e >= E) return;
  int d = dst[e];
#pragma unroll
  for (int h = 0; h < HEADS; ++h) {
    float ex = expf(alpha[e * HEADS + h] - amax[d * HEADS + h]);
    alpha[e * HEADS + h] = ex;
    atomicAdd(denom + d * HEADS + h, ex);
  }
}

// out[dst, dim] += (1/3) * sum_h H[src, h*64+dim] * (ex / (denom[dst,h]+1e-16))
// one wave per edge; lane handles dims {lane, lane+32}
__global__ void edge_scatter_kernel(const float* __restrict__ H,
                                    const int* __restrict__ src,
                                    const int* __restrict__ dst,
                                    const float* __restrict__ alpha,
                                    const float* __restrict__ denom,
                                    float* __restrict__ out,      // (N,64)
                                    int E) {
  int e    = (blockIdx.x * blockDim.x + threadIdx.x) >> 5;
  int lane = threadIdx.x & 31;
  if (e >= E) return;
  int s = src[e], d = dst[e];
  float w[HEADS];
#pragma unroll
  for (int h = 0; h < HEADS; ++h)
    w[h] = alpha[e * HEADS + h] / (denom[d * HEADS + h] + 1e-16f);
  const float* Hs = H + (size_t)s * FEAT;
#pragma unroll
  for (int j = 0; j < 2; ++j) {
    int dim = lane + j * 32;
    float acc = 0.0f;
#pragma unroll
    for (int h = 0; h < HEADS; ++h)
      acc += Hs[h * HID + dim] * w[h];
    atomicAdd(out + (size_t)d * HID + dim, acc * (1.0f / 3.0f));
  }
}

// out = (relu?) (out + bias[col])
__global__ void bias_act_kernel(float* __restrict__ out,
                                const float* __restrict__ bias,
                                int total, int D, int do_relu) {
  int i = blockIdx.x * blockDim.x + threadIdx.x;
  if (i >= total) return;
  float v = out[i] + bias[i % D];
  out[i] = do_relu ? fmaxf(v, 0.0f) : v;
}

// log_softmax over 40 cols, one wave per row, in place
__global__ void log_softmax40_kernel(float* __restrict__ O, int N) {
  int row  = (blockIdx.x * blockDim.x + threadIdx.x) >> 5;
  int lane = threadIdx.x & 31;
  if (row >= N) return;
  float* r = O + (size_t)row * 40;
  float v0 = r[lane];
  float v1 = (lane < 8) ? r[32 + lane] : -INFINITY;
  float m = fmaxf(v0, v1);
#pragma unroll
  for (int off = 16; off; off >>= 1) m = fmaxf(m, __shfl_xor(m, off, 32));
  float s = expf(v0 - m) + ((lane < 8) ? expf(v1 - m) : 0.0f);
#pragma unroll
  for (int off = 16; off; off >>= 1) s += __shfl_xor(s, off, 32);
  float lse = m + logf(s);
  r[lane] = v0 - lse;
  if (lane < 8) r[32 + lane] = v1 - lse;
}

// ------------------------------------------------------------------------------
static inline int cdiv(long a, long b) { return (int)((a + b - 1) / b); }

extern "C" void kernel_launch(void* const* d_in, const int* in_sizes, int n_in,
                              void* d_out, int out_size, void* d_ws, size_t ws_size,
                              hipStream_t stream) {
  const float* x     = (const float*)d_in[0];
  const int*   eidx  = (const int*)  d_in[1];   // (2, E)
  // d_in[2] = batch (unused)
  const float* W1    = (const float*)d_in[3];
  const float* b1    = (const float*)d_in[4];
  const float* att1  = (const float*)d_in[5];
  const float* bias1 = (const float*)d_in[6];
  const float* W2    = (const float*)d_in[7];
  const float* b2    = (const float*)d_in[8];
  const float* att2  = (const float*)d_in[9];
  const float* bias2 = (const float*)d_in[10];
  const float* pW1   = (const float*)d_in[11];
  const float* pb1   = (const float*)d_in[12];
  const float* pW2   = (const float*)d_in[13];
  const float* pb2   = (const float*)d_in[14];

  const int N = in_sizes[0] / 128;   // 50000
  const int E = in_sizes[1] / 2;     // 800000
  const int* src = eidx;
  const int* dst = eidx + E;

  // workspace layout (floats)
  float*    H     = (float*)d_ws;              // N*192
  float*    OUT   = H + (size_t)N * FEAT;      // N*64
  float*    ALPHA = OUT + (size_t)N * HID;     // E*3
  unsigned* AMAXU = (unsigned*)(ALPHA + (size_t)E * HEADS);  // N*3
  float*    DENOM = (float*)AMAXU + (size_t)N * HEADS;       // N*3

  const int BLK = 256;
  float* out40 = (float*)d_out;

  auto gemm_grid = [&](int M, int Nc) {
    long waves = (long)cdiv(M, 16) * cdiv(Nc, 64);
    return cdiv(waves * 32, BLK);
  };

  auto gat_attention = [&](const float* att, const float* bias) {
    fill0_kernel<<<cdiv((long)N * HEADS, BLK), BLK, 0, stream>>>((float*)AMAXU, N * HEADS);
    fill0_kernel<<<cdiv((long)N * HEADS, BLK), BLK, 0, stream>>>(DENOM, N * HEADS);
    fill0_kernel<<<cdiv((long)N * HID, BLK), BLK, 0, stream>>>(OUT, N * HID);
    edge_alpha_kernel<<<cdiv(E, BLK), BLK, 0, stream>>>(H, src, dst, att,
                                                        ALPHA, AMAXU, E);
    amax_decode_kernel<<<cdiv((long)N * HEADS, BLK), BLK, 0, stream>>>(AMAXU, N * HEADS);
    edge_exp_kernel<<<cdiv(E, BLK), BLK, 0, stream>>>((float*)AMAXU, dst,
                                                      ALPHA, DENOM, E);
    edge_scatter_kernel<<<cdiv((long)E * 32, BLK), BLK, 0, stream>>>(H, src, dst,
                                                                     ALPHA, DENOM,
                                                                     OUT, E);
    bias_act_kernel<<<cdiv((long)N * HID, BLK), BLK, 0, stream>>>(OUT, bias,
                                                                  N * HID, HID, 1);
  };

  // ---- layer 1 ----
  wmma_gemm_bias<128, 192><<<gemm_grid(N, 192), BLK, 0, stream>>>(x, W1, b1, H, N, 0);
  gat_attention(att1, bias1);                 // OUT = relu(gat) (N,64)

  // ---- layer 2 ----
  wmma_gemm_bias<64, 192><<<gemm_grid(N, 192), BLK, 0, stream>>>(OUT, W2, b2, H, N, 0);
  gat_attention(att2, bias2);                 // OUT = relu(gat) (N,64)

  // ---- MLP head ----
  wmma_gemm_bias<64, 64><<<gemm_grid(N, 64), BLK, 0, stream>>>(OUT, pW1, pb1, H, N, 0);
  wmma_gemm_bias<64, 40><<<gemm_grid(N, 40), BLK, 0, stream>>>(H, pW2, pb2, out40, N, 0);

  // ---- log_softmax in place ----
  log_softmax40_kernel<<<cdiv((long)N * 32, BLK), BLK, 0, stream>>>(out40, N);
}
